// SimpleAttention_79594333929721
// MI455X (gfx1250) — matrix operand order repaired and checked by reading, not compiled, not run on hardware
//
#include <hip/hip_runtime.h>

typedef __attribute__((ext_vector_type(16))) __bf16 v16bf;
typedef __attribute__((ext_vector_type(8)))  __bf16 v8bf;
typedef __attribute__((ext_vector_type(8)))  float  v8f;
typedef __attribute__((ext_vector_type(4)))  unsigned int v4u;
typedef __attribute__((ext_vector_type(8)))  int    v8i;
typedef __attribute__((ext_vector_type(4)))  int    v4i;

#define BN 4
#define LN 16384
#define DN 256
#define HN 8
#define DKN 32

// Build a 16-element bf16 fragment from two contiguous 16B runs in memory/LDS.
static __device__ __forceinline__ v16bf frag_from(const __bf16* lo, const __bf16* hi) {
    v8bf a = *(const v8bf*)lo;
    v8bf b = *(const v8bf*)hi;
    v16bf r;
#pragma unroll
    for (int i = 0; i < 8; ++i) { r[i] = a[i]; r[i + 8] = b[i]; }
    return r;
}

__global__ void k_zero(float* __restrict__ p, int n) {
    for (int i = blockIdx.x * blockDim.x + threadIdx.x; i < n; i += gridDim.x * blockDim.x)
        p[i] = 0.0f;
}

// ---------------------------------------------------------------------------
// Kernel 1: S_b = X_b^T X_b (fp32 atomic accumulation), sx_b = column sums,
// and xb16 = bf16(x) side-copy (written by jslab==0 blocks only).
// Grid: (32 L-blocks, 4 column slabs of 64, B). Block: 256 threads = 8 waves.
// Each wave: 2 row tiles x 4 col tiles (64 accum VGPRs -> no spills). LDS
// holds a 64-token chunk of X, bf16, transposed [dim][token].
// ---------------------------------------------------------------------------
#define TSTR 72   // padded token stride (elements); rows 144B, 16B-aligned
__global__ void __launch_bounds__(256, 1)
k_xtx(const float* __restrict__ x, float* __restrict__ S,
      float* __restrict__ sx, __bf16* __restrict__ xb16) {
    __shared__ __bf16 xt[DN * TSTR];
    const int b = blockIdx.z, jslab = blockIdx.y, lblk = blockIdx.x;
    const int tid = threadIdx.x, wave = tid >> 5, lane = tid & 31;
    const int lm = lane & 15, hi = lane >> 4;

    v8f acc[2][4];
    v8f vz = {};
#pragma unroll
    for (int ri = 0; ri < 2; ++ri)
#pragma unroll
        for (int jt = 0; jt < 4; ++jt) acc[ri][jt] = vz;

    float sxacc = 0.0f;
    const float* xb = x + (size_t)b * LN * DN;

    for (int t = lblk; t < LN / 64; t += 32) {
        const int tok0 = t * 64;
        for (int e = tid; e < 64 * DN; e += 256) {
            const int tok = e >> 8, dim = e & 255;
            const __bf16 bvv = (__bf16)xb[(size_t)(tok0 + tok) * DN + dim];
            xt[dim * TSTR + tok] = bvv;
            if (jslab == 0)
                xb16[((size_t)b * LN + tok0 + tok) * DN + dim] = bvv;
        }
        __syncthreads();

        if (jslab == 0) {
#pragma unroll 8
            for (int k = 0; k < 64; ++k) sxacc += (float)xt[tid * TSTR + k];
        }

#pragma unroll
        for (int kk = 0; kk < 2; ++kk) {
            const int k0 = kk * 32;
            v16bf afr[2];
#pragma unroll
            for (int ri = 0; ri < 2; ++ri) {
                const int row = (wave * 2 + ri) * 16 + lm;   // dim index (M)
                const __bf16* base = &xt[row * TSTR + k0 + hi * 8];
                afr[ri] = frag_from(base, base + 16);        // K 0..7/16..23 | 8..15/24..31
            }
#pragma unroll
            for (int jt = 0; jt < 4; ++jt) {
                const int col = jslab * 64 + jt * 16 + lm;   // dim index (N)
                const __bf16* base = &xt[col * TSTR + k0 + hi * 8];
                v16bf bfr = frag_from(base, base + 16);      // K 0..7/16..23 | 8..15/24..31
#pragma unroll
                for (int ri = 0; ri < 2; ++ri)
                    acc[ri][jt] = __builtin_amdgcn_wmma_f32_16x16x32_bf16(
                        false, afr[ri], false, bfr, (short)0, acc[ri][jt], false, false);
            }
        }
        __syncthreads();
    }

    float* Sb = S + (size_t)b * DN * DN;
#pragma unroll
    for (int ri = 0; ri < 2; ++ri)
#pragma unroll
        for (int jt = 0; jt < 4; ++jt)
#pragma unroll
            for (int e = 0; e < 8; ++e) {
                const int row = (wave * 2 + ri) * 16 + e + hi * 8;  // M = e + 8*(lane>=16)
                const int col = jslab * 64 + jt * 16 + lm;          // N = lane%16
                atomicAdd(&Sb[row * DN + col], acc[ri][jt][e]);
            }
    if (jslab == 0) atomicAdd(&sx[b * DN + tid], sxacc);
}

// ---------------------------------------------------------------------------
// Kernel 2a: p_h = (Wk_h S Wv_h^T + Wk_h sx bv^T + bk (Wv_h sx)^T + L bk bv^T)/L
// ---------------------------------------------------------------------------
__global__ void k_pattn(const float* __restrict__ S, const float* __restrict__ sx,
                        const float* __restrict__ Wk, const float* __restrict__ bk,
                        const float* __restrict__ Wv, const float* __restrict__ bv,
                        float* __restrict__ p) {
    __shared__ float wvU[DKN * DN];   // first Wv_h [e][j], then reused as U [i][e]
    __shared__ float svE[DKN];
    __shared__ float sxs[DN];
    const int b = blockIdx.x >> 3, h = blockIdx.x & 7;
    const int tid = threadIdx.x;

    for (int e = tid; e < DKN * DN; e += 256) wvU[e] = Wv[(size_t)(h * DKN) * DN + e];
    sxs[tid] = sx[b * DN + tid];
    __syncthreads();

    float uacc[DKN];
#pragma unroll
    for (int e = 0; e < DKN; ++e) uacc[e] = 0.0f;
    const float* Srow = S + ((size_t)b * DN + tid) * DN;
    for (int j = 0; j < DN; ++j) {
        const float s = Srow[j];
#pragma unroll
        for (int e = 0; e < DKN; ++e) uacc[e] += s * wvU[e * DN + j];
    }
    if (tid < DKN) {
        float a = 0.0f;
        for (int j = 0; j < DN; ++j) a += wvU[tid * DN + j] * sxs[j];
        svE[tid] = a;
    }
    __syncthreads();
#pragma unroll
    for (int e = 0; e < DKN; ++e) wvU[tid * DKN + e] = uacc[e];  // now U[i*32+e]
    __syncthreads();

    const int d = tid >> 3, e0 = (tid & 7) * 4;
    float acc4[4] = {0.0f, 0.0f, 0.0f, 0.0f};
    float wkdot = 0.0f;
    const float* wkr = Wk + (size_t)(h * DKN + d) * DN;
    for (int i = 0; i < DN; ++i) {
        const float w = wkr[i];
        wkdot += w * sxs[i];
#pragma unroll
        for (int t = 0; t < 4; ++t) acc4[t] += w * wvU[i * DKN + e0 + t];
    }
    const float bkd = bk[h * DKN + d];
    float* pp = p + (size_t)(b * HN + h) * DKN * DKN;
#pragma unroll
    for (int t = 0; t < 4; ++t) {
        const int e = e0 + t;
        const float bve = bv[h * DKN + e];
        pp[d * DKN + e] =
            (acc4[t] + wkdot * bve + bkd * (svE[e] + (float)LN * bve)) * (1.0f / (float)LN);
    }
}

// ---------------------------------------------------------------------------
// Kernel 2b: M1 = blockdiag(p_h^T) * Wq ; m1b = blockdiag(p_h^T) * bq
// ---------------------------------------------------------------------------
__global__ void k_m1(const float* __restrict__ p, const float* __restrict__ Wq,
                     const float* __restrict__ bq, float* __restrict__ M1,
                     float* __restrict__ m1b) {
    __shared__ float ph[DKN * DKN];
    const int b = blockIdx.x >> 3, h = blockIdx.x & 7;
    const int tid = threadIdx.x;
    for (int e = tid; e < DKN * DKN; e += 256)
        ph[e] = p[(size_t)(b * HN + h) * DKN * DKN + e];
    __syncthreads();

    float acc[DKN];
#pragma unroll
    for (int e = 0; e < DKN; ++e) acc[e] = 0.0f;
    for (int d = 0; d < DKN; ++d) {
        const float w = Wq[(size_t)(h * DKN + d) * DN + tid];
#pragma unroll
        for (int e = 0; e < DKN; ++e) acc[e] += ph[d * DKN + e] * w;
    }
    float* M1b = M1 + (size_t)b * DN * DN;
#pragma unroll
    for (int e = 0; e < DKN; ++e) M1b[(h * DKN + e) * DN + tid] = acc[e];

    if (tid < DKN) {
        float s = 0.0f;
        for (int d = 0; d < DKN; ++d) s += ph[d * DKN + tid] * bq[h * DKN + d];
        m1b[b * DN + h * DKN + tid] = s;
    }
}

// ---------------------------------------------------------------------------
// Kernel 2c: Weff = Wfc * M1 (stored row-major bf16), beff = Wfc*m1b + bfc
// ---------------------------------------------------------------------------
__global__ void k_weff(const float* __restrict__ M1, const float* __restrict__ m1b,
                       const float* __restrict__ Wfc, const float* __restrict__ bfc,
                       __bf16* __restrict__ Weffb, float* __restrict__ beff) {
    __shared__ float wfc[16 * DN];
    const int b = blockIdx.x >> 4, r0 = (blockIdx.x & 15) * 16;
    const int tid = threadIdx.x;
    for (int e = tid; e < 16 * DN; e += 256)
        wfc[e] = Wfc[(size_t)(r0 + (e >> 8)) * DN + (e & 255)];
    __syncthreads();

    float acc[16];
#pragma unroll
    for (int r = 0; r < 16; ++r) acc[r] = 0.0f;
    const float* M1b = M1 + (size_t)b * DN * DN;
    for (int k = 0; k < DN; ++k) {
        const float m = M1b[k * DN + tid];
#pragma unroll
        for (int r = 0; r < 16; ++r) acc[r] += wfc[r * DN + k] * m;
    }
    __bf16* Wb = Weffb + (size_t)b * DN * DN;
#pragma unroll
    for (int r = 0; r < 16; ++r) Wb[(size_t)(r0 + r) * DN + tid] = (__bf16)acc[r];

    if (tid < 16) {
        float s = bfc[r0 + tid];
        for (int k = 0; k < DN; ++k) s += wfc[tid * DN + k] * m1b[b * DN + k];
        beff[b * DN + r0 + tid] = s;
    }
}

// ---------------------------------------------------------------------------
// Kernel 3: out = X @ Weff^T + beff.  Grid: (L/64, B). Block: 256 = 8 waves.
// The 64x256 bf16 X tile is staged into LDS by the Tensor Data Mover:
// D# with data_size=2B, tile 256x64, tensor_dim0_stride=256, pad_enable with
// pad_interval=6 (128 DW = 512B rows) + pad_amount=3 (4 DW = 16B) to produce
// the 264-element padded LDS row stride. Issued by wave 0; completion via
// s_wait_tensorcnt 0, then workgroup barrier.
// ---------------------------------------------------------------------------
#define XSTR 264  // 512B row + 16B TDM pad = 528B, 16B-aligned rows
__global__ void __launch_bounds__(256, 1)
k_apply(const __bf16* __restrict__ xb16, const __bf16* __restrict__ Weffb,
        const float* __restrict__ beff, float* __restrict__ out) {
    __shared__ __bf16 xs[64 * XSTR];   // only shared object -> LDS offset 0
    const int b = blockIdx.y, tok0 = blockIdx.x * 64;
    const int tid = threadIdx.x, wave = tid >> 5, lane = tid & 31;
    const int lm = lane & 15, hi = lane >> 4;

    if (wave == 0) {
        const unsigned long long ga =
            (unsigned long long)(const void*)(xb16 + ((size_t)b * LN + tok0) * DN);
        v4u g0;
        g0[0] = 1u;                                            // count=1, user mode
        g0[1] = 0u;                                            // lds_addr = 0 (xs)
        g0[2] = (unsigned int)(ga & 0xFFFFFFFFull);            // global_addr[31:0]
        g0[3] = (unsigned int)((ga >> 32) & 0x1FFFFFFull)      // global_addr[56:32]
              | (2u << 30);                                    // type = 2 (image)
        v8i g1;
        g1[0] = (1 << 16)      // data_size = 1 -> 2 bytes
              | (1 << 20)      // pad_enable
              | (6 << 22)      // pad_interval: 128 DW = 512B
              | (3 << 25);     // pad_amount: 4 DW = 16B
        g1[1] = (int)(256u << 16);   // tensor_dim0 = 256  (bits 79:48)
        g1[2] = (int)(64u << 16);    // tensor_dim1 = 64   (bits 111:80)
        g1[3] = (int)(256u << 16);   // tile_dim0 = 256    (bits 127:112)
        g1[4] = 64;                  // tile_dim1 = 64     (bits 143:128)
        g1[5] = 256;                 // tensor_dim0_stride = 256 (bits 207:160)
        g1[6] = 0;
        g1[7] = 0;
        v4i z4 = {0, 0, 0, 0};
#if __clang_major__ >= 23
        v8i z8 = {0, 0, 0, 0, 0, 0, 0, 0};
        __builtin_amdgcn_tensor_load_to_lds(g0, g1, z4, z4, z8, 0);
#else
        __builtin_amdgcn_tensor_load_to_lds(g0, g1, z4, z4, 0);
#endif
        __builtin_amdgcn_s_wait_tensorcnt(0);
    }
    __syncthreads();

    v8f acc[4][2];
    v8f vz = {};
#pragma unroll
    for (int ri = 0; ri < 4; ++ri)
#pragma unroll
        for (int rj = 0; rj < 2; ++rj) acc[ri][rj] = vz;

    const __bf16* Wb = Weffb + (size_t)b * DN * DN;
    for (int k0 = 0; k0 < DN; k0 += 32) {
        if (k0 + 32 < DN)  // warm caches for next B slab
            __builtin_prefetch(Wb + (size_t)((wave * 2) * 16 + lm) * DN + k0 + 32, 0, 1);

        v16bf afr[4], bfr[2];
#pragma unroll
        for (int ri = 0; ri < 4; ++ri) {
            const __bf16* base = &xs[(ri * 16 + lm) * XSTR + k0 + hi * 8];
            afr[ri] = frag_from(base, base + 16);
        }
#pragma unroll
        for (int rj = 0; rj < 2; ++rj) {
            const int col = (wave * 2 + rj) * 16 + lm;
            const __bf16* base = Wb + (size_t)col * DN + k0 + hi * 8;
            bfr[rj] = frag_from(base, base + 16);
        }
#pragma unroll
        for (int ri = 0; ri < 4; ++ri)
#pragma unroll
            for (int rj = 0; rj < 2; ++rj)
                acc[ri][rj] = __builtin_amdgcn_wmma_f32_16x16x32_bf16(
                    false, afr[ri], false, bfr[rj], (short)0, acc[ri][rj], false, false);
    }

    float* ob = out + ((size_t)b * LN + tok0) * DN;
#pragma unroll
    for (int rj = 0; rj < 2; ++rj) {
        const int col = (wave * 2 + rj) * 16 + lm;
        const float bvv = beff[b * DN + col];
#pragma unroll
        for (int ri = 0; ri < 4; ++ri)
#pragma unroll
            for (int e = 0; e < 8; ++e) {
                const int m = ri * 16 + e + hi * 8;   // M = e + 8*(lane>=16)
                ob[(size_t)m * DN + col] = acc[ri][rj][e] + bvv;
            }
    }
}

// ---------------------------------------------------------------------------
extern "C" void kernel_launch(void* const* d_in, const int* in_sizes, int n_in,
                              void* d_out, int out_size, void* d_ws, size_t ws_size,
                              hipStream_t stream) {
    (void)in_sizes; (void)n_in; (void)out_size; (void)ws_size;
    const float* x   = (const float*)d_in[0];
    const float* Wq  = (const float*)d_in[1];
    const float* bq  = (const float*)d_in[2];
    const float* Wk  = (const float*)d_in[3];
    const float* bk  = (const float*)d_in[4];
    const float* Wv  = (const float*)d_in[5];
    const float* bv  = (const float*)d_in[6];
    const float* Wfc = (const float*)d_in[7];
    const float* bfc = (const float*)d_in[8];
    float* out = (float*)d_out;

    // Workspace layout (floats, then bf16):
    float* wsf    = (float*)d_ws;
    float* S      = wsf;                        // B*D*D   = 262144
    float* sxp    = S + BN * DN * DN;           // B*D     = 1024
    float* pP     = sxp + BN * DN;              // B*H*DK*DK = 32768
    float* M1     = pP + BN * HN * DKN * DKN;   // B*D*D   = 262144
    float* m1b    = M1 + BN * DN * DN;          // B*D
    float* beff   = m1b + BN * DN;              // B*D
    __bf16* Weffb = (__bf16*)(beff + BN * DN);  // B*D*D bf16
    __bf16* xb16  = Weffb + (size_t)BN * DN * DN; // B*L*D bf16 = 32MB

    // 1. zero the atomic accumulators (S, sx are contiguous)
    k_zero<<<512, 256, 0, stream>>>(S, BN * DN * DN + BN * DN);
    // 2. S_b = X_b^T X_b, sx_b, and xb16 = bf16(x)
    k_xtx<<<dim3(32, 4, BN), 256, 0, stream>>>(x, S, sxp, xb16);
    // 3. per-(b,h) 32x32 p_attn with bias cross-terms
    k_pattn<<<BN * HN, 256, 0, stream>>>(S, sxp, Wk, bk, Wv, bv, pP);
    // 4. M1 = blockdiag(p^T) Wq, m1b = blockdiag(p^T) bq
    k_m1<<<BN * HN, 256, 0, stream>>>(pP, Wq, bq, M1, m1b);
    // 5. Weff = Wfc M1 (bf16), beff = Wfc m1b + bfc
    k_weff<<<BN * 16, 256, 0, stream>>>(M1, m1b, Wfc, bfc, Weffb, beff);
    // 6. out = X Weff^T + beff (X tile staged via Tensor Data Mover)
    k_apply<<<dim3(LN / 64, BN), 256, 0, stream>>>(xb16, Weffb, beff, out);
}
